// PerformerAttention_47090021433955
// MI455X (gfx1250) — matrix operand order repaired and checked by reading, not compiled, not run on hardware
//
#include <hip/hip_runtime.h>
#include <hip/hip_bf16.h>
#include <math.h>

typedef _Float16 f16;
typedef __attribute__((ext_vector_type(16))) _Float16 v16h;
typedef __attribute__((ext_vector_type(8)))  _Float16 v8h;
typedef __attribute__((ext_vector_type(8)))  float    v8f;

#define D_MODEL 512
#define N_TOK   2048
#define N_HEADS 8
#define DK      64
#define M_FEAT  266
#define MP      288     // feature dim padded to 9*32
#define CH      32      // chunk length (tokens)
#define NCH     (N_TOK / CH)   // 64 chunks
#define BC      80      // state cols: 64 V + z + z_total + 14 pad (5*16)
#define KEPS    1e-4f
#define NSTAB   1e-6f
#define LNEPS   1e-6f
#define QKSCALE 0.21022410381342865f   // 512^-0.25

// ---------------- WMMA helpers (gfx1250, wave32) ----------------

__device__ __forceinline__ v8f wmma_f16(v16h a, v16h b, v8f c) {
  // v_wmma_f32_16x16x32_f16: D = A(16x32) * B(32x16) + C(16x16 f32)
  return __builtin_amdgcn_wmma_f32_16x16x32_f16(
      false, a, false, b, (short)0, c, false, false);
}

// A operand 16x32 f16 from row-major storage (row stride lda elements).
// ISA layout: lane<16 holds row r=lane, K in {kb*8..kb*8+7, 16+kb*8..+7}
__device__ __forceinline__ v16h load_a(const f16* A, int lda) {
  int lane = threadIdx.x & 31;
  int r = lane & 15, kb = (lane >> 4) & 1;
  const f16* p = A + r * lda + kb * 8;
  v8h lo = *(const v8h*)p;
  v8h hi = *(const v8h*)(p + 16);
  v16h out;
#pragma unroll
  for (int i = 0; i < 8; ++i) { out[i] = lo[i]; out[8 + i] = hi[i]; }
  return out;
}

// A operand where logical A[r][k] = M[k*ldm + r]  (i.e. A = M^T), strided loads
__device__ __forceinline__ v16h load_a_tr(const f16* M, int ldm) {
  int lane = threadIdx.x & 31;
  int r = lane & 15, kb = (lane >> 4) & 1;
  v16h out;
#pragma unroll
  for (int i = 0; i < 8; ++i) {
    out[i]     = M[(kb * 8 + i) * ldm + r];
    out[8 + i] = M[(16 + kb * 8 + i) * ldm + r];
  }
  return out;
}

// B operand 32x16 f16 from row-major B^T storage BT[n][k] (row stride ld).
// Layout: lane<16 -> col n=lane, K in {kb*8..kb*8+7, 16+kb*8..+7}, kb=lane>>4
__device__ __forceinline__ v16h load_b(const f16* BT, int ld) {
  int lane = threadIdx.x & 31;
  int n = lane & 15, k0 = (lane >> 4) * 8;
  const f16* p = BT + n * ld + k0;
  v8h lo = *(const v8h*)p;
  v8h hi = *(const v8h*)(p + 16);
  v16h out;
#pragma unroll
  for (int i = 0; i < 8; ++i) { out[i] = lo[i]; out[8 + i] = hi[i]; }
  return out;
}

// B operand from row-major KxN storage M[k][n] (row stride ldm), strided loads
__device__ __forceinline__ v16h load_b_tr(const f16* M, int ldm) {
  int lane = threadIdx.x & 31;
  int n = lane & 15, k0 = (lane >> 4) * 8;
  v16h out;
#pragma unroll
  for (int i = 0; i < 8; ++i) { out[i] = M[(k0 + i) * ldm + n]; out[8 + i] = M[(16 + k0 + i) * ldm + n]; }
  return out;
}

// ---------------- prep kernels ----------------

__global__ void k_prep_w(const float* __restrict__ Wq, const float* __restrict__ Wk,
                         const float* __restrict__ Wv, const float* __restrict__ Wfc,
                         f16* WqT, f16* WkT, f16* WvT, f16* WfcT) {
  int idx = blockIdx.x * blockDim.x + threadIdx.x;   // 512*512
  int n = idx >> 9, k = idx & 511;
  const float* W = (blockIdx.y == 0) ? Wq : (blockIdx.y == 1) ? Wk
                   : (blockIdx.y == 2) ? Wv : Wfc;
  f16* WT = (blockIdx.y == 0) ? WqT : (blockIdx.y == 1) ? WkT
            : (blockIdx.y == 2) ? WvT : WfcT;
  WT[n * 512 + k] = (f16)W[k * 512 + n];             // B^T layout [n][k]
}

__global__ void k_prep_rf(const float* __restrict__ rf, f16* rf16) {
  int idx = blockIdx.x * blockDim.x + threadIdx.x;   // 288*64
  int m = idx >> 6, d = idx & 63;
  rf16[idx] = (m < M_FEAT) ? (f16)rf[m * 64 + d] : (f16)0.0f;
}

__global__ void k_init_stab(unsigned int* stab_bits) {
  *stab_bits = 0x7F800000u;  // +inf (we atomicMin non-negative 0.5*||k||^2)
}

__global__ void k_ln(const float* __restrict__ q, const float* __restrict__ gamma,
                     const float* __restrict__ beta, f16* qn) {
  __shared__ float red[256];
  int n = blockIdx.x, t = threadIdx.x;
  float a = q[n * 512 + t], b = q[n * 512 + 256 + t];
  red[t] = a + b;
  __syncthreads();
  for (int s = 128; s > 0; s >>= 1) { if (t < s) red[t] += red[t + s]; __syncthreads(); }
  float mu = red[0] * (1.0f / 512.0f);
  __syncthreads();
  float da = a - mu, db = b - mu;
  red[t] = da * da + db * db;
  __syncthreads();
  for (int s = 128; s > 0; s >>= 1) { if (t < s) red[t] += red[t + s]; __syncthreads(); }
  float inv = rsqrtf(red[0] * (1.0f / 512.0f) + LNEPS);
  qn[n * 512 + t]       = (f16)(da * inv * gamma[t] + beta[t]);
  qn[n * 512 + 256 + t] = (f16)(db * inv * gamma[t + 256] + beta[t + 256]);
}

__global__ void k_cvt(const float* __restrict__ k, const float* __restrict__ v,
                      f16* k16, f16* v16) {
  int idx = blockIdx.x * blockDim.x + threadIdx.x;   // 2048*512
  k16[idx] = (f16)k[idx];
  v16[idx] = (f16)v[idx];
}

// ---------------- QKV projection GEMM: [2048,512] x [512,512] ----------------

__global__ void __launch_bounds__(256)
k_qkv(const f16* __restrict__ qn, const f16* __restrict__ k16, const f16* __restrict__ v16,
      const f16* __restrict__ WqT, const f16* __restrict__ WkT, const f16* __restrict__ WvT,
      f16* qh, f16* kh, f16* vh) {
  int which = blockIdx.z;
  const f16* A  = (which == 0) ? qn  : (which == 1) ? k16 : v16;
  const f16* BT = (which == 0) ? WqT : (which == 1) ? WkT : WvT;
  f16* O        = (which == 0) ? qh  : (which == 1) ? kh  : vh;
  float scale   = (which == 2) ? 1.0f : QKSCALE;

  int wave = threadIdx.x >> 5;
  int tm = blockIdx.x;                 // 128 token tiles
  int tn = blockIdx.y * 8 + wave;      // 32 column tiles
  const f16* Ab = A + tm * 16 * 512;
  const f16* Bb = BT + tn * 16 * 512;
  v8f acc = {};
  for (int k0 = 0; k0 < 512; k0 += 32)
    acc = wmma_f16(load_a(Ab + k0, 512), load_b(Bb + k0, 512), acc);

  int lane = threadIdx.x & 31;
  int col = tn * 16 + (lane & 15);
  int r0  = tm * 16 + (lane >> 4) * 8;
  int h = col >> 6, d = col & 63;      // head-major storage [h][n][64]
#pragma unroll
  for (int j = 0; j < 8; ++j)
    O[(h * N_TOK + (r0 + j)) * DK + d] = (f16)(acc[j] * scale);
}

// ---------------- h_k and global key stabilizer ----------------

__global__ void k_hk(const f16* __restrict__ kh, float* hk, unsigned int* stab_bits) {
  __shared__ float red[256];
  int idx = blockIdx.x * blockDim.x + threadIdx.x;   // 8*2048
  const f16* p = kh + idx * DK;
  float s = 0.0f;
#pragma unroll
  for (int d = 0; d < DK; ++d) { float x = (float)p[d]; s += x * x; }
  s *= 0.5f;                       // = -h_k, non-negative
  hk[idx] = -s;
  red[threadIdx.x] = s;
  __syncthreads();
  for (int st = 128; st > 0; st >>= 1) {
    if (threadIdx.x < st) red[threadIdx.x] = fminf(red[threadIdx.x], red[threadIdx.x + st]);
    __syncthreads();
  }
  if (threadIdx.x == 0)
    atomicMin(stab_bits, __float_as_uint(red[0]));   // bit-order == float-order for >=0
}

// ---------------- feature map GEMM + exp epilogue ----------------

__global__ void __launch_bounds__(256)
k_feat(const f16* __restrict__ qh, const f16* __restrict__ kh, const f16* __restrict__ rf16,
       const float* __restrict__ hk, const unsigned int* __restrict__ stab_bits,
       f16* qp, f16* kp) {
  int isK = blockIdx.z;
  int h = blockIdx.y;
  int id = blockIdx.x * 8 + (threadIdx.x >> 5);   // 0..2303
  int tm = id & 127;                               // token tile
  int tn = id >> 7;                                // feature tile 0..17
  const f16* A  = (isK ? kh : qh) + (h * N_TOK + tm * 16) * DK;
  const f16* BT = rf16 + tn * 16 * DK;
  v8f acc = {};
  for (int k0 = 0; k0 < DK; k0 += 32)
    acc = wmma_f16(load_a(A + k0, DK), load_b(BT + k0, DK), acc);

  float kstab = -__uint_as_float(*stab_bits);
  const float cfac = rsqrtf((float)M_FEAT);
  int lane = threadIdx.x & 31;
  int m  = tn * 16 + (lane & 15);
  int r0 = tm * 16 + (lane >> 4) * 8;
  f16* O = (isK ? kp : qp) + (size_t)h * N_TOK * MP;
#pragma unroll
  for (int j = 0; j < 8; ++j) {
    int n = r0 + j;
    float e = acc[j];
    if (isK) e += hk[h * N_TOK + n] - kstab;   // q: h_q cancels exactly
    float val = (m < M_FEAT) ? cfac * (expf(e) + KEPS) : 0.0f;
    O[(size_t)n * MP + m] = (f16)val;
  }
}

// ---------------- pass 1: per-chunk state S_c = K'^T (288x32) x [V|1|0] (32x80) ----

__global__ void __launch_bounds__(256)
k_pass1(const f16* __restrict__ kp, const f16* __restrict__ vh, f16* cs) {
  int h = blockIdx.y, c = blockIdx.x;
  __shared__ __align__(16) f16 ks[CH * MP];     // K' chunk [tok][m]
  __shared__ __align__(16) f16 vbT[BC * CH];    // Vb^T [col][tok]
  const f16* src = kp + ((size_t)h * N_TOK + c * CH) * MP;
  for (int i = threadIdx.x; i < CH * MP; i += 256) ks[i] = src[i];
  const f16* vsrc = vh + ((size_t)h * N_TOK + c * CH) * DK;
  for (int i = threadIdx.x; i < BC * CH; i += 256) {
    int col = i >> 5, t = i & 31;
    vbT[i] = (col < 64) ? vsrc[t * DK + col] : (col == 64 ? (f16)1.0f : (f16)0.0f);
  }
  __syncthreads();

  int wave = threadIdx.x >> 5, lane = threadIdx.x & 31;
  f16* out = cs + ((size_t)(h * NCH + c)) * MP * BC;
  for (int t = wave; t < 18 * 5; t += 8) {      // 18 m-tiles x 5 col-tiles
    int tm = t / 5, tn = t % 5;
    v16h a = load_a_tr(ks + tm * 16, MP);       // A[m][tok] = K'[tok][m]
    v16h b = load_b(vbT + tn * 16 * CH, CH);
    v8f acc = {};
    acc = wmma_f16(a, b, acc);
    int col = tn * 16 + (lane & 15), r0 = tm * 16 + (lane >> 4) * 8;
#pragma unroll
    for (int j = 0; j < 8; ++j) out[(r0 + j) * BC + col] = (f16)acc[j];
  }
}

// ---------------- pass 2: exclusive prefix over chunks + z_total column ----------

__global__ void k_pass2(const f16* __restrict__ cs, f16* Bc) {
  int idx = blockIdx.x * blockDim.x + threadIdx.x;  // 8*288*80 = 184320
  if (idx >= N_HEADS * MP * BC) return;
  int col = idx % BC;
  int m   = (idx / BC) % MP;
  int h   = idx / (BC * MP);
  size_t base = ((size_t)h * NCH) * MP * BC + (size_t)m * BC + col;
  size_t step = (size_t)MP * BC;
  if (col < 65) {                                   // V cols + z col
    float run = 0.0f;
    for (int c = 0; c < NCH; ++c) {
      float cur = (float)cs[base + c * step];
      Bc[base + c * step] = (f16)run;               // exclusive prefix
      run += cur;
    }
    if (col == 64)                                  // col 65 := z_total (all chunks)
      for (int c = 0; c < NCH; ++c) Bc[base + 1 + c * step] = (f16)run;
  } else if (col >= 66) {                           // zero padding cols
    for (int c = 0; c < NCH; ++c) Bc[base + c * step] = (f16)0.0f;
  }
}

// ---------------- pass 3: Out = Q'*[S|z|ztot] + mask(Q'K'^T)*[V|1|0], normalize ----

__global__ void __launch_bounds__(256)
k_pass3(const f16* __restrict__ qp, const f16* __restrict__ kp,
        const f16* __restrict__ vh, const f16* __restrict__ Bc, f16* ctx) {
  int h = blockIdx.y, c = blockIdx.x;
  __shared__ __align__(16) f16  qs[CH * MP];
  __shared__ __align__(16) f16  ks[CH * MP];
  __shared__ __align__(16) f16  vbT[BC * CH];
  __shared__ __align__(16) float aqk[CH * CH];
  __shared__ __align__(16) f16  am[CH * CH];
  __shared__ __align__(16) float ob[CH * BC];

  const f16* qsrc = qp + ((size_t)h * N_TOK + c * CH) * MP;
  const f16* ksrc = kp + ((size_t)h * N_TOK + c * CH) * MP;
  for (int i = threadIdx.x; i < CH * MP; i += 256) { qs[i] = qsrc[i]; ks[i] = ksrc[i]; }
  const f16* vsrc = vh + ((size_t)h * N_TOK + c * CH) * DK;
  for (int i = threadIdx.x; i < BC * CH; i += 256) {
    int col = i >> 5, t = i & 31;
    vbT[i] = (col < 64) ? vsrc[t * DK + col] : (col == 64 ? (f16)1.0f : (f16)0.0f);
  }
  __syncthreads();

  int wave = threadIdx.x >> 5, lane = threadIdx.x & 31;

  // A_qk = Q' (32x288) x K'^T (288x32): 4 tiles on waves 0..3
  if (wave < 4) {
    int tm = wave >> 1, tn = wave & 1;
    v8f acc = {};
    for (int k0 = 0; k0 < MP; k0 += 32) {
      v16h a = load_a(qs + tm * 16 * MP + k0, MP);
      v16h b = load_b(ks + tn * 16 * MP + k0, MP);  // B^T row n = K' row n
      acc = wmma_f16(a, b, acc);
    }
    int col = tn * 16 + (lane & 15), r0 = tm * 16 + (lane >> 4) * 8;
#pragma unroll
    for (int j = 0; j < 8; ++j) aqk[(r0 + j) * CH + col] = acc[j];
  }
  __syncthreads();
  for (int i = threadIdx.x; i < CH * CH; i += 256) {
    int row = i / CH, colj = i % CH;
    am[i] = (colj <= row) ? (f16)aqk[i] : (f16)0.0f; // causal incl. diagonal
  }
  __syncthreads();

  // Out (32x80): inter (K=288) + intra (K=32); 10 tiles over 8 waves
  const f16* Bbase = Bc + ((size_t)(h * NCH + c)) * MP * BC;
  for (int t = wave; t < 10; t += 8) {
    int tm = t / 5, tn = t % 5;
    v8f acc = {};
    for (int k0 = 0; k0 < MP; k0 += 32) {
      v16h a = load_a(qs + tm * 16 * MP + k0, MP);
      v16h b = load_b_tr(Bbase + (size_t)k0 * BC + tn * 16, BC);
      acc = wmma_f16(a, b, acc);
    }
    v16h a2 = load_a(am + tm * 16 * CH, CH);
    v16h b2 = load_b(vbT + tn * 16 * CH, CH);
    acc = wmma_f16(a2, b2, acc);
    int col = tn * 16 + (lane & 15), r0 = tm * 16 + (lane >> 4) * 8;
#pragma unroll
    for (int j = 0; j < 8; ++j) ob[(r0 + j) * BC + col] = acc[j];
  }
  __syncthreads();

  // epilogue: col 64 = q'.cumsum(k') (causal), col 65 = q'.sum(k') (global)
  for (int i = threadIdx.x; i < CH * DK; i += 256) {
    int t = i >> 6, e = i & 63;
    float Dn = ob[t * BC + 64];
    float d  = ob[t * BC + 65];
    if (fabsf(d) <= NSTAB) d += 2.0f * NSTAB;
    float y = ob[t * BC + e] / (Dn * d);
    ctx[((size_t)(c * CH + t)) * D_MODEL + h * DK + e] = (f16)y;
  }
}

// ---------------- output GEMM + bias + residual ----------------

__global__ void __launch_bounds__(256)
k_out(const f16* __restrict__ ctx, const f16* __restrict__ WfcT,
      const float* __restrict__ bfc, const float* __restrict__ q, float* out) {
  int wave = threadIdx.x >> 5;
  int tm = blockIdx.x;
  int tn = blockIdx.y * 8 + wave;
  const f16* Ab = ctx + tm * 16 * 512;
  const f16* Bb = WfcT + tn * 16 * 512;
  v8f acc = {};
  for (int k0 = 0; k0 < 512; k0 += 32)
    acc = wmma_f16(load_a(Ab + k0, 512), load_b(Bb + k0, 512), acc);
  int lane = threadIdx.x & 31;
  int col = tn * 16 + (lane & 15), r0 = tm * 16 + (lane >> 4) * 8;
  float bias = bfc[col];
#pragma unroll
  for (int j = 0; j < 8; ++j) {
    int n = r0 + j;
    out[n * 512 + col] = acc[j] + bias + q[n * 512 + col];
  }
}

// ---------------- host launcher ----------------

static size_t alloc_off(size_t& off, size_t bytes) {
  size_t r = off;
  off += (bytes + 255) & ~(size_t)255;
  return r;
}

extern "C" void kernel_launch(void* const* d_in, const int* in_sizes, int n_in,
                              void* d_out, int out_size, void* d_ws, size_t ws_size,
                              hipStream_t stream) {
  const float* q     = (const float*)d_in[0];
  const float* k     = (const float*)d_in[1];
  const float* v     = (const float*)d_in[2];
  const float* Wq    = (const float*)d_in[3];
  const float* Wk    = (const float*)d_in[4];
  const float* Wv    = (const float*)d_in[5];
  const float* Wfc   = (const float*)d_in[6];
  const float* bfc   = (const float*)d_in[7];
  const float* gamma = (const float*)d_in[8];
  const float* beta  = (const float*)d_in[9];
  const float* rf    = (const float*)d_in[10];
  float* out = (float*)d_out;
  char* ws = (char*)d_ws;

  size_t off = 0;
  f16* qn16  = (f16*)(ws + alloc_off(off, (size_t)N_TOK * 512 * 2));
  f16* k16   = (f16*)(ws + alloc_off(off, (size_t)N_TOK * 512 * 2));
  f16* v16   = (f16*)(ws + alloc_off(off, (size_t)N_TOK * 512 * 2));
  f16* WqT   = (f16*)(ws + alloc_off(off, (size_t)512 * 512 * 2));
  f16* WkT   = (f16*)(ws + alloc_off(off, (size_t)512 * 512 * 2));
  f16* WvT   = (f16*)(ws + alloc_off(off, (size_t)512 * 512 * 2));
  f16* WfcT  = (f16*)(ws + alloc_off(off, (size_t)512 * 512 * 2));
  f16* rf16  = (f16*)(ws + alloc_off(off, (size_t)MP * DK * 2));
  f16* qh16  = (f16*)(ws + alloc_off(off, (size_t)N_HEADS * N_TOK * DK * 2));
  f16* kh16  = (f16*)(ws + alloc_off(off, (size_t)N_HEADS * N_TOK * DK * 2));
  f16* vh16  = (f16*)(ws + alloc_off(off, (size_t)N_HEADS * N_TOK * DK * 2));
  float* hk  = (float*)(ws + alloc_off(off, (size_t)N_HEADS * N_TOK * 4));
  unsigned int* stab = (unsigned int*)(ws + alloc_off(off, 256));
  f16* qp16  = (f16*)(ws + alloc_off(off, (size_t)N_HEADS * N_TOK * MP * 2));
  f16* kp16  = (f16*)(ws + alloc_off(off, (size_t)N_HEADS * N_TOK * MP * 2));
  f16* cs16  = (f16*)(ws + alloc_off(off, (size_t)N_HEADS * NCH * MP * BC * 2));
  f16* Bc16  = (f16*)(ws + alloc_off(off, (size_t)N_HEADS * NCH * MP * BC * 2));
  f16* ctx16 = (f16*)(ws + alloc_off(off, (size_t)N_TOK * 512 * 2));
  (void)ws_size; (void)in_sizes; (void)n_in; (void)out_size;

  k_prep_w<<<dim3(512 * 512 / 256, 4), 256, 0, stream>>>(Wq, Wk, Wv, Wfc,
                                                         WqT, WkT, WvT, WfcT);
  k_prep_rf<<<MP * DK / 256, 256, 0, stream>>>(rf, rf16);
  k_init_stab<<<1, 1, 0, stream>>>(stab);
  k_ln<<<N_TOK, 256, 0, stream>>>(q, gamma, beta, qn16);
  k_cvt<<<N_TOK * 512 / 256, 256, 0, stream>>>(k, v, k16, v16);

  k_qkv<<<dim3(N_TOK / 16, 4, 3), 256, 0, stream>>>(qn16, k16, v16,
                                                    WqT, WkT, WvT,
                                                    qh16, kh16, vh16);
  k_hk<<<N_HEADS * N_TOK / 256, 256, 0, stream>>>(kh16, hk, stab);
  k_feat<<<dim3(288, N_HEADS, 2), 256, 0, stream>>>(qh16, kh16, rf16, hk, stab,
                                                    qp16, kp16);
  k_pass1<<<dim3(NCH, N_HEADS), 256, 0, stream>>>(kp16, vh16, cs16);
  k_pass2<<<(N_HEADS * MP * BC + 255) / 256, 256, 0, stream>>>(cs16, Bc16);
  k_pass3<<<dim3(NCH, N_HEADS), 256, 0, stream>>>(qp16, kp16, vh16, Bc16, ctx16);
  k_out<<<dim3(N_TOK / 16, 4), 256, 0, stream>>>(ctx16, WfcT, bfc, q, out);
}
